// TBStars2MoESparseBlock_23278722744751
// MI455X (gfx1250) — compile-verified, matrix operand order checked by reading, not run on hardware
//
#include <hip/hip_runtime.h>
#include <cstdint>
#include <cstddef>

#define TT 4096
#define HH 1024
#define FF 2048
#define EE 8
#define KK 2
#define XSTR 1032   // LDS row stride in halfs (16B multiple, padded vs 1024 to dodge bank conflicts)
#define MB 32       // tokens per block (two 16-row WMMA M-tiles)

typedef __attribute__((ext_vector_type(16))) __bf16          v16bf;
typedef __attribute__((ext_vector_type(8)))  float           v8f;
typedef __attribute__((ext_vector_type(4)))  unsigned int    u32x4;

union ABPack { u32x4 q[2]; v16bf b; };

__device__ __forceinline__ unsigned short f2bf(float f) {
  unsigned int u = __float_as_uint(f);
  u += 0x7FFFu + ((u >> 16) & 1u);          // round-to-nearest-even
  return (unsigned short)(u >> 16);
}

__device__ __forceinline__ float silu(float g) {
  // g * sigmoid(g) with fast v_rcp_f32 instead of the IEEE divide sequence
  return g * __builtin_amdgcn_rcpf(1.f + __expf(-g));
}

// ---------------------------------------------------------------- zero init
__global__ void moe_zero(float* __restrict__ out, int* __restrict__ counts) {
  size_t i = (size_t)blockIdx.x * 256 + threadIdx.x;   // one float4 per thread
  float4 z; z.x = z.y = z.z = z.w = 0.f;
  ((float4*)out)[i] = z;
  if (blockIdx.x == 0 && threadIdx.x < EE) counts[threadIdx.x] = 0;
}

// ------------------------------------------------- f32 -> bf16 weight pass
__global__ void moe_cvt(const float* __restrict__ w1, const float* __restrict__ w2,
                        unsigned short* __restrict__ w1b, unsigned short* __restrict__ w2b) {
  const size_t N1 = (size_t)EE * 2 * FF * HH;
  size_t i = ((size_t)blockIdx.x * 256 + threadIdx.x) * 8;
  const float* src; unsigned short* dst;
  if (i < N1) { src = w1 + i;        dst = w1b + i; }
  else        { src = w2 + (i - N1); dst = w2b + (i - N1); }
  float4 a = *(const float4*)src;
  float4 b = *(const float4*)(src + 4);
  union { unsigned short s[8]; u32x4 q; } pk;
  pk.s[0] = f2bf(a.x); pk.s[1] = f2bf(a.y); pk.s[2] = f2bf(a.z); pk.s[3] = f2bf(a.w);
  pk.s[4] = f2bf(b.x); pk.s[5] = f2bf(b.y); pk.s[6] = f2bf(b.z); pk.s[7] = f2bf(b.w);
  *(u32x4*)dst = pk.q;
}

// ------------------------------------------------------------------ router
// one thread per (token, expert); 8-lane groups live inside one wave32
__global__ void moe_router(const float* __restrict__ hs, const float* __restrict__ gw,
                           float* __restrict__ logits_out, int* __restrict__ counts,
                           int* __restrict__ entries, float* __restrict__ wgts) {
  int tid = blockIdx.x * blockDim.x + threadIdx.x;
  int t = tid >> 3;
  int e = tid & 7;
  if (t >= TT) return;
  const float4* xr = (const float4*)(hs + (size_t)t * HH);
  const float4* gr = (const float4*)(gw + (size_t)e * HH);
  float acc = 0.f;
#pragma unroll 4
  for (int i = 0; i < HH / 4; ++i) {
    float4 a = xr[i], b = gr[i];
    acc += a.x * b.x + a.y * b.y + a.z * b.z + a.w * b.w;
  }
  logits_out[(size_t)t * EE + e] = acc;

  // softmax over the 8-lane expert group
  float mx = acc;
  for (int m = 4; m >= 1; m >>= 1) mx = fmaxf(mx, __shfl_xor(mx, m, 8));
  float p = __expf(acc - mx);
  float s = p;
  for (int m = 4; m >= 1; m >>= 1) s += __shfl_xor(s, m, 8);
  p /= s;

  // top-1 (tie -> lowest expert index, matching lax.top_k)
  float v1 = p; int i1 = e;
  for (int m = 4; m >= 1; m >>= 1) {
    float ov = __shfl_xor(v1, m, 8);
    int   oi = __shfl_xor(i1, m, 8);
    if (ov > v1 || (ov == v1 && oi < i1)) { v1 = ov; i1 = oi; }
  }
  // top-2
  float v2 = (e == i1) ? -1.f : p; int i2 = e;
  for (int m = 4; m >= 1; m >>= 1) {
    float ov = __shfl_xor(v2, m, 8);
    int   oi = __shfl_xor(i2, m, 8);
    if (ov > v2 || (ov == v2 && oi < i2)) { v2 = ov; i2 = oi; }
  }
  float inv = 1.f / (v1 + v2);
  if (e == i1) {
    int pos = atomicAdd(&counts[e], 1);
    entries[e * TT + pos] = (t << 1);
    wgts[e * TT + pos] = v1 * inv;
  } else if (e == i2) {
    int pos = atomicAdd(&counts[e], 1);
    entries[e * TT + pos] = (t << 1) | 1;
    wgts[e * TT + pos] = v2 * inv;
  }
}

// -------------------------------------------------- GEMM1: x @ w1^T, SiLU*up
// block = 256 threads (8 waves), one 32-token tile (two WMMA M-tiles) of one
// expert's list. Each wave owns one 16-wide N column pair (gate+up) and runs
// 4 WMMAs per k-step off a single B load -> weight bytes amortized 2x.
__global__ void moe_gemm1(const float* __restrict__ hs, const unsigned short* __restrict__ w1b,
                          const int* __restrict__ counts, const int* __restrict__ entries,
                          unsigned short* __restrict__ act) {
  const int MT = TT / MB;
  int e = blockIdx.x / MT, mt = blockIdx.x % MT;
  int cnt = counts[e];
  int base = mt * MB;
  if (base >= cnt) return;

  __shared__ int s_tok[MB], s_slot[MB];
  __shared__ __align__(16) unsigned short s_x[MB * XSTR];   // 66 KB (<320 KB/WGP)

  int tid = threadIdx.x;
  if (tid < MB) {
    int p = base + tid;
    int ent = (p < cnt) ? entries[e * TT + p] : 0;
    s_tok[tid]  = (p < cnt) ? (ent >> 1) : 0;
    s_slot[tid] = (p < cnt) ? ent : (TT * KK + e * MB + tid);   // dummy pad slot
  }
  __syncthreads();

  // cooperative gather of MB x-rows into LDS (f32 -> bf16)
  for (int c = tid; c < MB * (HH / 4); c += 256) {
    int row = c >> 8;          // HH/4 == 256
    int c4  = c & 255;
    float4 v = *(const float4*)(hs + (size_t)s_tok[row] * HH + c4 * 4);
    union { unsigned short s[4]; unsigned long long q; } pk;
    pk.s[0] = f2bf(v.x); pk.s[1] = f2bf(v.y); pk.s[2] = f2bf(v.z); pk.s[3] = f2bf(v.w);
    *(unsigned long long*)&s_x[row * XSTR + c4 * 4] = pk.q;
  }
  __syncthreads();

  int wave = tid >> 5, lane = tid & 31;
  int m    = lane & 15;
  int koff = (lane >> 4) * 8;     // lanes 16-31 hold K 8..15 / 24..31

  for (int nt = wave; nt < FF / 16; nt += 8) {
    const unsigned short* pg = w1b + ((size_t)e * 2 * FF + nt * 16 + (lane & 15)) * HH;
    const unsigned short* pu = w1b + ((size_t)e * 2 * FF + FF + nt * 16 + (lane & 15)) * HH;
    v8f cg0 = {}; v8f cu0 = {}; v8f cg1 = {}; v8f cu1 = {};
    for (int k0 = 0; k0 < HH; k0 += 32) {
      ABPack a0, a1, bg, bu;
      a0.q[0] = *(const u32x4*)&s_x[m * XSTR + k0 + koff];
      a0.q[1] = *(const u32x4*)&s_x[m * XSTR + k0 + 16 + koff];
      a1.q[0] = *(const u32x4*)&s_x[(16 + m) * XSTR + k0 + koff];
      a1.q[1] = *(const u32x4*)&s_x[(16 + m) * XSTR + k0 + 16 + koff];
      bg.q[0] = *(const u32x4*)(pg + k0 + koff);
      bg.q[1] = *(const u32x4*)(pg + k0 + 16 + koff);
      bu.q[0] = *(const u32x4*)(pu + k0 + koff);
      bu.q[1] = *(const u32x4*)(pu + k0 + 16 + koff);
      cg0 = __builtin_amdgcn_wmma_f32_16x16x32_bf16(false, a0.b, false, bg.b, (short)0, cg0, false, false);
      cu0 = __builtin_amdgcn_wmma_f32_16x16x32_bf16(false, a0.b, false, bu.b, (short)0, cu0, false, false);
      cg1 = __builtin_amdgcn_wmma_f32_16x16x32_bf16(false, a1.b, false, bg.b, (short)0, cg1, false, false);
      cu1 = __builtin_amdgcn_wmma_f32_16x16x32_bf16(false, a1.b, false, bu.b, (short)0, cu1, false, false);
    }
    int nb = nt * 16 + (lane & 15);
    int mb = (lane >> 4) * 8;
#pragma unroll
    for (int r = 0; r < 8; ++r) {
      act[(size_t)s_slot[mb + r]      * FF + nb] = f2bf(silu(cg0[r]) * cu0[r]);
      act[(size_t)s_slot[16 + mb + r] * FF + nb] = f2bf(silu(cg1[r]) * cu1[r]);
    }
  }
}

// --------------------------------------- GEMM2: act @ w2^T, weighted scatter
__global__ void moe_gemm2(const unsigned short* __restrict__ act, const unsigned short* __restrict__ w2b,
                          const int* __restrict__ counts, const int* __restrict__ entries,
                          const float* __restrict__ wgts, float* __restrict__ out) {
  const int MT = TT / MB;
  int e = blockIdx.x / MT, mt = blockIdx.x % MT;
  int cnt = counts[e];
  int base = mt * MB;
  if (base >= cnt) return;

  __shared__ int   s_tok[MB], s_slot[MB];
  __shared__ float s_w[MB];
  int tid = threadIdx.x;
  if (tid < MB) {
    int p = base + tid;
    int ent = (p < cnt) ? entries[e * TT + p] : 0;
    s_tok[tid]  = ent >> 1;
    s_slot[tid] = (p < cnt) ? ent : (TT * KK + e * MB + tid);
    s_w[tid]    = (p < cnt) ? wgts[e * TT + p] : 0.f;
  }
  __syncthreads();

  int wave = tid >> 5, lane = tid & 31;
  int m    = lane & 15;
  int koff = (lane >> 4) * 8;
  const unsigned short* arow0 = act + (size_t)s_slot[m]      * FF;
  const unsigned short* arow1 = act + (size_t)s_slot[16 + m] * FF;

  for (int nt = wave; nt < HH / 16; nt += 8) {
    const unsigned short* pb = w2b + ((size_t)e * HH + nt * 16 + (lane & 15)) * FF;
    v8f c0 = {}; v8f c1 = {};
    for (int k0 = 0; k0 < FF; k0 += 32) {
      ABPack a0, a1, bp;
      a0.q[0] = *(const u32x4*)(arow0 + k0 + koff);
      a0.q[1] = *(const u32x4*)(arow0 + k0 + 16 + koff);
      a1.q[0] = *(const u32x4*)(arow1 + k0 + koff);
      a1.q[1] = *(const u32x4*)(arow1 + k0 + 16 + koff);
      bp.q[0] = *(const u32x4*)(pb + k0 + koff);
      bp.q[1] = *(const u32x4*)(pb + k0 + 16 + koff);
      c0 = __builtin_amdgcn_wmma_f32_16x16x32_bf16(false, a0.b, false, bp.b, (short)0, c0, false, false);
      c1 = __builtin_amdgcn_wmma_f32_16x16x32_bf16(false, a1.b, false, bp.b, (short)0, c1, false, false);
    }
    int nb = nt * 16 + (lane & 15);
    int mb = (lane >> 4) * 8;
#pragma unroll
    for (int r = 0; r < 8; ++r) {
      int m0 = mb + r, m1 = 16 + mb + r;
      atomicAdd(out + (size_t)s_tok[m0] * HH + nb, s_w[m0] * c0[r]);
      atomicAdd(out + (size_t)s_tok[m1] * HH + nb, s_w[m1] * c1[r]);
    }
  }
}

// ------------------------------------------------------------------ launch
extern "C" void kernel_launch(void* const* d_in, const int* in_sizes, int n_in,
                              void* d_out, int out_size, void* d_ws, size_t ws_size,
                              hipStream_t stream) {
  (void)in_sizes; (void)n_in; (void)out_size; (void)ws_size;
  const float* hs = (const float*)d_in[0];
  const float* gw = (const float*)d_in[1];
  const float* w1 = (const float*)d_in[2];
  const float* w2 = (const float*)d_in[3];
  float* out = (float*)d_out;

  char* ws = (char*)d_ws;
  int*   counts  = (int*)ws;                                  // 8 ints (256B reserved)
  int*   entries = (int*)(ws + 256);                          // [E*T]
  float* wgts    = (float*)(ws + 256 + (size_t)EE * TT * 4);  // [E*T]
  unsigned short* act = (unsigned short*)(ws + 256 + 2 * (size_t)EE * TT * 4);
  size_t actBytes = (size_t)(TT * KK + EE * MB) * FF * 2;     // ~34 MB (+pad slots)
  unsigned short* w1b = (unsigned short*)((char*)act + actBytes);   // 67 MB
  unsigned short* w2b = w1b + (size_t)EE * 2 * FF * HH;             // 33.5 MB

  moe_zero  <<<(TT * HH / 4) / 256, 256, 0, stream>>>(out, counts);
  moe_cvt   <<<((size_t)EE * 2 * FF * HH + (size_t)EE * HH * FF) / 8 / 256, 256, 0, stream>>>(w1, w2, w1b, w2b);
  moe_router<<<(TT * EE) / 256, 256, 0, stream>>>(hs, gw, out + (size_t)TT * HH, counts, entries, wgts);
  moe_gemm1 <<<EE * (TT / MB), 256, 0, stream>>>(hs, w1b, counts, entries, act);
  moe_gemm2 <<<EE * (TT / MB), 256, 0, stream>>>(act, w2b, counts, entries, wgts, out);
}
